// Param2MoEMoEBlock_29076928594669
// MI455X (gfx1250) — matrix operand order repaired and checked by reading, not compiled, not run on hardware
//
#include <hip/hip_runtime.h>
#include <hip/hip_bf16.h>

// ---------------------------------------------------------------------------
// DeepSeek-style MoE block for MI455X (gfx1250, wave32, WMMA).
//   T=2048 tokens, H=1024 hidden, E=64 experts, I=704, TOPK=6, NG=8, TG=4
// Strategy: router -> per-expert token buckets -> grouped GEMM with
// v_wmma_f32_16x16x32_bf16 (fp32 weights converted to bf16 in-flight),
// fp32-accumulate, SwiGLU fused, fp32 atomic combine. Shared expert is
// appended as expert #64 over all tokens with weight 1.
//
// Round-2: M-tile doubled to 32 tokens per workgroup (two 16-row A fragments
// share every B fragment) -> halves L2->WGP weight traffic and vmem
// instructions per FLOP. Wave id stays scalarized (readfirstlane) so all tile
// guards are s_cbranch_scc and EXEC is all-1s across every WMMA.
// ---------------------------------------------------------------------------

#define T_    2048
#define H_    1024
#define E_    64
#define I_    704
#define TWOI  1408
#define TOPK_ 6
#define NG_   8
#define TG_   4
#define RSF_  2.5f
#define CAP_  2048          // worst-case tokens per expert
#define AS_   712           // LDS row stride for activation tile (bf16)
#define MT_   32            // tokens per workgroup (two 16-row WMMA tiles)

typedef __bf16 bf16t;
typedef __attribute__((ext_vector_type(16))) __bf16 v16bf;
typedef __attribute__((ext_vector_type(8)))  float  v8f;

__device__ __forceinline__ v8f wmma_bf16(v16bf a, v16bf b, v8f c) {
  // args: (neg_a, A, neg_b, B, c_mod, C, reuse_a, reuse_b)
  return __builtin_amdgcn_wmma_f32_16x16x32_bf16(false, a, false, b,
                                                 (short)0, c, false, false);
}

__device__ __forceinline__ v8f zero8() {
  v8f z;
#pragma unroll
  for (int i = 0; i < 8; ++i) z[i] = 0.f;
  return z;
}

// ---------------------------------------------------------------------------
// Zero output + per-expert counters (d_out/d_ws are poisoned by the harness).
// ---------------------------------------------------------------------------
__global__ void zero_kernel(float* __restrict__ out, unsigned* __restrict__ cnt) {
  size_t i = (size_t)blockIdx.x * 256 + threadIdx.x;
  if (i < (size_t)T_ * H_) out[i] = 0.f;
  if (i < E_) cnt[i] = 0u;
}

// ---------------------------------------------------------------------------
// Router: logits = x @ gate_w^T, sigmoid scores, grouped top-k selection,
// renormalized weights * RSF, scatter into per-expert buckets.
// One 64-thread block (2 waves) per token; lane e owns expert e's dot product.
// ---------------------------------------------------------------------------
__global__ __launch_bounds__(64) void router_kernel(
    const float* __restrict__ x, const float* __restrict__ gw,
    const float* __restrict__ bias, unsigned* __restrict__ cnt,
    int* __restrict__ btok, float* __restrict__ bwt) {
  __shared__ float xs[H_];
  __shared__ float logit[E_];
  const int t = blockIdx.x, tid = threadIdx.x;

  const float* xr = x + (size_t)t * H_;
  for (int i = tid; i < H_; i += 64) xs[i] = xr[i];
  __syncthreads();

  const float* g = gw + (size_t)tid * H_;
  float acc = 0.f;
  for (int h = 0; h < H_; h += 4)
    acc += xs[h] * g[h] + xs[h + 1] * g[h + 1] +
           xs[h + 2] * g[h + 2] + xs[h + 3] * g[h + 3];
  logit[tid] = acc;
  __syncthreads();

  if (tid == 0) {
    float sco[E_], sc[E_];
    for (int e = 0; e < E_; ++e) {
      float s = 1.f / (1.f + __expf(-logit[e]));
      sco[e] = s;
      sc[e]  = s + bias[e];
    }
    // group score = sum of top-2 biased scores within each group of 8
    float gs[NG_];
    for (int gi = 0; gi < NG_; ++gi) {
      float m1 = -3.4e38f, m2 = -3.4e38f;
      for (int j = 0; j < E_ / NG_; ++j) {
        float v = sc[gi * (E_ / NG_) + j];
        if (v > m1) { m2 = m1; m1 = v; } else if (v > m2) { m2 = v; }
      }
      gs[gi] = m1 + m2;
    }
    // top TG_ groups (first index wins ties, matching jax top_k)
    unsigned gmask = 0;
    for (int it = 0; it < TG_; ++it) {
      float best = -3.4e38f; int bi = 0;
      for (int gi = 0; gi < NG_; ++gi)
        if (!((gmask >> gi) & 1u) && gs[gi] > best) { best = gs[gi]; bi = gi; }
      gmask |= 1u << bi;
    }
    // top TOPK_ experts within selected groups
    unsigned long long esel = 0ull;
    int ids[TOPK_]; float wv[TOPK_]; float wsum = 0.f;
    for (int k = 0; k < TOPK_; ++k) {
      float best = -3.4e38f; int bi = 0;
      for (int e = 0; e < E_; ++e)
        if (((gmask >> (e >> 3)) & 1u) && !((esel >> e) & 1ull) && sc[e] > best) {
          best = sc[e]; bi = e;
        }
      esel |= 1ull << bi;
      ids[k] = bi; wv[k] = sco[bi]; wsum += sco[bi];
    }
    const float scale = RSF_ / wsum;
    for (int k = 0; k < TOPK_; ++k) {
      int e = ids[k];
      unsigned pos = atomicAdd(&cnt[e], 1u);
      btok[e * CAP_ + pos] = t;
      bwt [e * CAP_ + pos] = wv[k] * scale;
    }
  }
}

// ---------------------------------------------------------------------------
// Grouped expert GEMM. Grid = E_*64 routed (expert, 32-token tile) blocks
// plus 64 shared-expert tiles. Block = 256 threads = 8 waves.
// GEMM1: [32 x 1024] @ [1024 x 1408]; wave w owns gate tiles jg=w+8p and the
//        *matching* up tiles jg+44, so SwiGLU stays in-register per wave.
//        Two 16-row A fragments reuse every B fragment (4 WMMAs per B pair).
// GEMM2: [32 x 704] @ [704 x 1024]; activations exchanged via 45.6KB LDS.
// Combine: fp32 global atomic add scaled by routing weight (0 for pad rows).
// ---------------------------------------------------------------------------
__global__ __launch_bounds__(256) void moe_kernel(
    const float* __restrict__ x,
    const float* __restrict__ wgu_all, const float* __restrict__ wd_all,
    const float* __restrict__ wgu_sh,  const float* __restrict__ wd_sh,
    const unsigned* __restrict__ cnt, const int* __restrict__ btok,
    const float* __restrict__ bwt, float* __restrict__ out) {
  __shared__ int   tokrow[MT_];
  __shared__ float wrow[MT_];
  __shared__ bf16t act[MT_ * AS_];    // 32x704 bf16 activation tile (padded)

  const int bx = blockIdx.x;
  int e, m;
  const float *wgu, *wd;
  if (bx < E_ * (T_ / MT_)) {
    e = bx / (T_ / MT_); m = bx % (T_ / MT_);
    if ((unsigned)(m * MT_) >= cnt[e]) return;    // block-uniform early exit
    wgu = wgu_all + (size_t)e * H_ * TWOI;
    wd  = wd_all  + (size_t)e * I_ * H_;
  } else {
    e = E_; m = bx - E_ * (T_ / MT_);             // shared expert
    wgu = wgu_sh; wd = wd_sh;
  }

  const int tid = threadIdx.x;
  if (tid < MT_) {
    if (e < E_) {
      int r = m * MT_ + tid;
      if ((unsigned)r < cnt[e]) { tokrow[tid] = btok[e * CAP_ + r];
                                  wrow[tid]   = bwt [e * CAP_ + r]; }
      else                      { tokrow[tid] = 0; wrow[tid] = 0.f; }
    } else { tokrow[tid] = m * MT_ + tid; wrow[tid] = 1.f; }
  }
  __syncthreads();

  const int lane   = tid & 31;
  // Wave id as an explicit SGPR value: tile guards below become *scalar*
  // branches, so EXEC stays all-1s across every WMMA (ISA requirement).
  const int w      = __builtin_amdgcn_readfirstlane(tid >> 5);   // 0..7
  const int lane15 = lane & 15;
  const int half   = lane >> 4;      // lane half selects K sub-range (ISA layout)
  const float* xrow0 = x + (size_t)tokrow[lane15]      * H_;   // rows 0..15
  const float* xrow1 = x + (size_t)tokrow[16 + lane15] * H_;   // rows 16..31

  // ---------------- GEMM1 + in-register SwiGLU -----------------------------
  v8f aG0[6], aU0[6], aG1[6], aU1[6];
#pragma unroll
  for (int p = 0; p < 6; ++p) {
    aG0[p] = zero8(); aU0[p] = zero8();
    aG1[p] = zero8(); aU1[p] = zero8();
  }

  for (int k0 = 0; k0 < H_; k0 += 32) {
    // A fragments: 16-bit A 16x32 — lane half picks K {0..7,16..23} / {8..15,24..31}
    v16bf a0, a1;
    const float* ax0 = xrow0 + k0 + half * 8;
    const float* ax1 = xrow1 + k0 + half * 8;
#pragma unroll
    for (int j = 0; j < 8; ++j) {
      a0[j] = (bf16t)ax0[j];  a0[j + 8] = (bf16t)ax0[16 + j];
      a1[j] = (bf16t)ax1[j];  a1[j + 8] = (bf16t)ax1[16 + j];
    }
#pragma unroll
    for (int p = 0; p < 6; ++p) {
      const int jg = w + 8 * p;                  // gate column tile 0..43
      if (jg < 44) {                             // scalar (SGPR) guard
        const float* bg = wgu + (size_t)(k0 + half * 8) * TWOI + (jg * 16 + lane15);
        const float* bu = bg + I_;               // matching up column
        v16bf bG, bU;
#pragma unroll
        for (int i = 0; i < 16; ++i) {           // B: N=lane%16, K=half*8+i (+8 for i>=8)
          bG[i] = (bf16t)bg[(size_t)(i + (i & 8)) * TWOI];
          bU[i] = (bf16t)bu[(size_t)(i + (i & 8)) * TWOI];
        }
        aG0[p] = wmma_bf16(a0, bG, aG0[p]);
        aG1[p] = wmma_bf16(a1, bG, aG1[p]);
        aU0[p] = wmma_bf16(a0, bU, aU0[p]);
        aU1[p] = wmma_bf16(a1, bU, aU1[p]);
      }
    }
  }

  // SwiGLU: silu(g)*u, write bf16 activation tile to LDS (C layout M=r+8*half)
#pragma unroll
  for (int p = 0; p < 6; ++p) {
    const int jg = w + 8 * p;
    if (jg < 44) {                               // scalar guard
#pragma unroll
      for (int r = 0; r < 8; ++r) {
        const int Mr = r + 8 * half;
        float g0 = aG0[p][r], u0 = aU0[p][r];
        float g1 = aG1[p][r], u1 = aU1[p][r];
        float v0 = g0 / (1.f + __expf(-g0)) * u0;
        float v1 = g1 / (1.f + __expf(-g1)) * u1;
        act[Mr * AS_        + jg * 16 + lane15] = (bf16t)v0;
        act[(16 + Mr) * AS_ + jg * 16 + lane15] = (bf16t)v1;
      }
    }
  }
  __syncthreads();

  // ---------------- GEMM2: [32 x 704] @ [704 x 1024] -----------------------
  v8f c0[8], c1[8];
#pragma unroll
  for (int q = 0; q < 8; ++q) { c0[q] = zero8(); c1[q] = zero8(); }

  for (int k0 = 0; k0 < I_; k0 += 32) {
    v16bf a0, a1;
    const bf16t* ap0 = &act[lane15 * AS_        + k0 + half * 8];
    const bf16t* ap1 = &act[(16 + lane15) * AS_ + k0 + half * 8];
#pragma unroll
    for (int j = 0; j < 8; ++j) {
      a0[j] = ap0[j];  a0[j + 8] = ap0[16 + j];
      a1[j] = ap1[j];  a1[j + 8] = ap1[16 + j];
    }
#pragma unroll
    for (int q = 0; q < 8; ++q) {
      const int n2 = (w + 8 * q) * 16;           // 64 N tiles / 8 waves
      const float* bp = wd + (size_t)(k0 + half * 8) * H_ + (n2 + lane15);
      v16bf b;
#pragma unroll
      for (int i = 0; i < 16; ++i) b[i] = (bf16t)bp[(size_t)(i + (i & 8)) * H_];
      c0[q] = wmma_bf16(a0, b, c0[q]);
      c1[q] = wmma_bf16(a1, b, c1[q]);
    }
  }

  // ---------------- combine: out[t] += c_t * y  (pad rows have c=0) --------
#pragma unroll
  for (int q = 0; q < 8; ++q) {
    const int n2 = (w + 8 * q) * 16 + lane15;
#pragma unroll
    for (int r = 0; r < 8; ++r) {
      const int Mr = r + 8 * half;
      float v0 = c0[q][r] * wrow[Mr];
      float v1 = c1[q][r] * wrow[16 + Mr];
      unsafeAtomicAdd(&out[(size_t)tokrow[Mr] * H_ + n2], v0);
      unsafeAtomicAdd(&out[(size_t)tokrow[16 + Mr] * H_ + n2], v1);
    }
  }
}

// ---------------------------------------------------------------------------
extern "C" void kernel_launch(void* const* d_in, const int* in_sizes, int n_in,
                              void* d_out, int out_size, void* d_ws, size_t ws_size,
                              hipStream_t stream) {
  const float* x    = (const float*)d_in[0];   // [T, H]
  const float* gw   = (const float*)d_in[1];   // [E, H]
  const float* bias = (const float*)d_in[2];   // [E]
  const float* wgu  = (const float*)d_in[3];   // [E, H, 2I]
  const float* wd   = (const float*)d_in[4];   // [E, I, H]
  const float* wgus = (const float*)d_in[5];   // [H, 2I]
  const float* wds  = (const float*)d_in[6];   // [I, H]
  float* out = (float*)d_out;                  // [T, H] fp32

  // workspace: counts | per-expert token buckets | per-expert weight buckets
  unsigned* cnt  = (unsigned*)d_ws;
  int*      btok = (int*)  ((char*)d_ws + 256);
  float*    bwt  = (float*)((char*)d_ws + 256 + (size_t)E_ * CAP_ * sizeof(int));

  zero_kernel<<<(T_ * H_ + 255) / 256, 256, 0, stream>>>(out, cnt);
  router_kernel<<<T_, 64, 0, stream>>>(x, gw, bias, cnt, btok, bwt);
  moe_kernel<<<E_ * (T_ / MT_) + (T_ / MT_), 256, 0, stream>>>(
      x, wgu, wd, wgus, wds, cnt, btok, bwt, out);
}
